// Encoder_Decoder_35631048687667
// MI455X (gfx1250) — compile-verified, matrix operand order checked by reading, not run on hardware
//
#include <hip/hip_runtime.h>
#include <hip/hip_bf16.h>

// ---------------------------------------------------------------------------
// Batch-1 seq2seq (LSTM encoder/decoder + dot attention) for MI455X / gfx1250.
// All matvecs/GEMMs via v_wmma_f32_16x16x32_bf16 (bf16 weights, f32 accum).
// Weights converted to bf16 once per launch (~167MB ws; decoder set ~90MB fits
// the 192MB L2). Activations keep a bf16 shadow copy written by their producer
// so the WMMA inner loop is pure 16-byte bf16 loads + WMMA (no converts).
// Encoder input projection hoisted into one WMMA GEMM over all 128 timesteps.
// ---------------------------------------------------------------------------

typedef __attribute__((ext_vector_type(16))) __bf16 v16bf;
typedef __attribute__((ext_vector_type(8)))  __bf16 v8bf;
typedef __attribute__((ext_vector_type(8)))  float  v8f;

#define HDIM   1024
#define H4DIM  4096
#define SLEN   128
#define VOUT   32000
#define NSTEPS 51
#define EOS_ID 2

__device__ __forceinline__ float sigmoidf_(float x) { return 1.0f / (1.0f + expf(-x)); }

// 16-bf16 fragment for one lane from a row-major bf16 row.
// Layout (ISA 7.12.2, 16-bit 16x32): lane<16 holds K {0..7,16..23},
// lane>=16 holds K {8..15,24..31}.
__device__ __forceinline__ v16bf load_frag_row(const __bf16* row, int k, int off) {
    v8bf lo = *(const v8bf*)(row + k + off);
    v8bf hi = *(const v8bf*)(row + k + 16 + off);
    v16bf a;
#pragma unroll
    for (int i = 0; i < 8; ++i) { a[i] = lo[i]; a[8 + i] = hi[i]; }
    return a;
}

__device__ __forceinline__ v8f mv_accum(const __bf16* wrow, const __bf16* xbf,
                                        int K, int off, v8f acc) {
    for (int k = 0; k < K; k += 32) {
        __builtin_prefetch(wrow + k + 256, 0, 3);   // stream weights ahead (global_prefetch_b8)
        v16bf a = load_frag_row(wrow, k, off);
        v16bf b = load_frag_row(xbf,  k, off);      // x broadcast across all 16 N columns
        acc = __builtin_amdgcn_wmma_f32_16x16x32_bf16(false, a, false, b,
                                                      (short)0, acc, false, false);
    }
    return acc;
}

// y[M] = act( W1[MxK]@x1 (+ W2[MxK]@x2) (+b1)(+b2)(+addv) ), one wave per 16 rows.
// x vectors are pre-converted bf16. Optional bf16 shadow copy of y in ybf.
__global__ void wmma_matvec_kernel(const __bf16* __restrict__ W1, const __bf16* __restrict__ x1,
                                   const __bf16* __restrict__ W2, const __bf16* __restrict__ x2,
                                   const float* __restrict__ b1, const float* __restrict__ b2,
                                   const float* __restrict__ addv,
                                   float* __restrict__ y, __bf16* __restrict__ ybf,
                                   int M, int K, int act) {
    int lane = threadIdx.x & 31;
    int wv   = blockIdx.x * (blockDim.x >> 5) + (threadIdx.x >> 5);
    int r0   = wv << 4;
    if (r0 >= M) return;                       // wave-uniform: EXEC all-1 at WMMA
    int half = lane >> 4, off = half << 3, mrow = lane & 15;

    v8f acc = {0.f, 0.f, 0.f, 0.f, 0.f, 0.f, 0.f, 0.f};
    acc = mv_accum(W1 + (size_t)(r0 + mrow) * K, x1, K, off, acc);
    if (W2) acc = mv_accum(W2 + (size_t)(r0 + mrow) * K, x2, K, off, acc);

    // D 16x16 f32: lane0 holds y[r0..r0+7], lane16 holds y[r0+8..r0+15] (N=0 col)
    if (mrow == 0) {
        int mb = r0 + (half << 3);
#pragma unroll
        for (int v = 0; v < 8; ++v) {
            float r = acc[v];
            if (b1)   r += b1[mb + v];
            if (b2)   r += b2[mb + v];
            if (addv) r += addv[mb + v];
            if (act)  r = tanhf(r);
            y[mb + v] = r;
            if (ybf) ybf[mb + v] = (__bf16)r;
        }
    }
}

// C[n*M + m] = sum_k A[m][k] * B[n][k]   (A:[M,K] bf16, B:[N,K] bf16, C:[N,M] f32)
// One wave per 16x16 output tile. Encoder input projection (M=4096,N=128,K=1024).
__global__ void wmma_gemm_nt_kernel(const __bf16* __restrict__ A, const __bf16* __restrict__ B,
                                    float* __restrict__ C, int M, int N, int K) {
    int lane = threadIdx.x & 31;
    int wv   = blockIdx.x * (blockDim.x >> 5) + (threadIdx.x >> 5);
    int mtiles = M >> 4;
    int mt = wv % mtiles, nt = wv / mtiles;
    if (nt >= (N >> 4)) return;
    int r0 = mt << 4, n0 = nt << 4;
    int half = lane >> 4, off = half << 3, lrow = lane & 15;

    const __bf16* arow = A + (size_t)(r0 + lrow) * K;
    const __bf16* brow = B + (size_t)(n0 + lrow) * K;
    v8f acc = {0.f, 0.f, 0.f, 0.f, 0.f, 0.f, 0.f, 0.f};
    for (int k = 0; k < K; k += 32) {
        v16bf a = load_frag_row(arow, k, off);
        v16bf b = load_frag_row(brow, k, off);
        acc = __builtin_amdgcn_wmma_f32_16x16x32_bf16(false, a, false, b,
                                                      (short)0, acc, false, false);
    }
    int n = n0 + lrow;
    int mb = r0 + (half << 3);
#pragma unroll
    for (int v = 0; v < 8; ++v) C[(size_t)n * M + mb + v] = acc[v];
}

__global__ void cvt_f32_bf16_kernel(const float* __restrict__ s, __bf16* __restrict__ d, size_t n) {
    size_t i = (size_t)blockIdx.x * blockDim.x + threadIdx.x;
    size_t stride = (size_t)gridDim.x * blockDim.x;
    for (; i < n; i += stride) d[i] = (__bf16)s[i];
}

__global__ void embed_src_kernel(const int* __restrict__ ids, const float* __restrict__ emb,
                                 __bf16* __restrict__ X) {
    int t = blockIdx.x;
    const float* e = emb + (size_t)ids[t] * HDIM;
    for (int h = threadIdx.x; h < HDIM; h += blockDim.x)
        X[(size_t)t * HDIM + h] = (__bf16)e[h];
}

// bf16 gather of one embedding row selected by a device-side token id.
__global__ void gather_row_bf16_kernel(const float* __restrict__ table, const int* __restrict__ idx,
                                       __bf16* __restrict__ out) {
    int j = blockIdx.x * blockDim.x + threadIdx.x;
    if (j < HDIM) out[j] = (__bf16)table[(size_t)idx[0] * HDIM + j];
}

__global__ void init_state_kernel(float* __restrict__ h, float* __restrict__ c,
                                  __bf16* __restrict__ hbf) {
    int i = blockIdx.x * blockDim.x + threadIdx.x;
    if (i < HDIM) { h[i] = 0.f; c[i] = 0.f; hbf[i] = (__bf16)0.f; }
}

// gate order i,f,g,o (torch LSTMCell). blockDim = 1024.
__global__ void lstm_cell_kernel(const float* __restrict__ g, float* __restrict__ h,
                                 float* __restrict__ c, __bf16* __restrict__ hbf,
                                 float* __restrict__ hs_out) {
    int j = threadIdx.x;
    float ii = sigmoidf_(g[j]);
    float ff = sigmoidf_(g[HDIM + j]);
    float gg = tanhf(g[2 * HDIM + j]);
    float oo = sigmoidf_(g[3 * HDIM + j]);
    float c2 = ff * c[j] + ii * gg;
    c[j] = c2;
    float h2 = oo * tanhf(c2);
    h[j] = h2;
    hbf[j] = (__bf16)h2;
    if (hs_out) hs_out[j] = h2;
}

// scores = hs_all@ht; softmax; ctx = a@hs_all; concat = [ctx, ht] (+bf16 copy).
// blockDim = 1024; 8 threads cooperate per score.
__global__ void attn_kernel(const float* __restrict__ hs_all, const float* __restrict__ ht,
                            float* __restrict__ concat, __bf16* __restrict__ concat_bf) {
    __shared__ float part[1024];
    __shared__ float sc[SLEN];
    __shared__ float red[2];
    int tid = threadIdx.x;
    {   // scores: s = tid>>3, each of 8 threads sums 128 elements
        int s = tid >> 3, p = tid & 7;
        const float* row = hs_all + (size_t)s * HDIM + p * 128;
        const float* hp  = ht + p * 128;
        float acc = 0.f;
        for (int k = 0; k < 128; ++k) acc += row[k] * hp[k];
        part[tid] = acc;
    }
    __syncthreads();
    if (tid < SLEN) {
        float acc = 0.f;
#pragma unroll
        for (int p = 0; p < 8; ++p) acc += part[tid * 8 + p];
        sc[tid] = acc;
    }
    __syncthreads();
    if (tid == 0) { float m = sc[0]; for (int s = 1; s < SLEN; ++s) m = fmaxf(m, sc[s]); red[0] = m; }
    __syncthreads();
    if (tid < SLEN) sc[tid] = expf(sc[tid] - red[0]);
    __syncthreads();
    if (tid == 0) { float sum = 0.f; for (int s = 0; s < SLEN; ++s) sum += sc[s]; red[1] = sum; }
    __syncthreads();
    float inv = 1.0f / red[1];
    float acc = 0.f;
    for (int s = 0; s < SLEN; ++s) acc += sc[s] * hs_all[(size_t)s * HDIM + tid];
    float d = acc * inv, h = ht[tid];
    concat[tid] = d;            concat_bf[tid] = (__bf16)d;
    concat[HDIM + tid] = h;     concat_bf[HDIM + tid] = (__bf16)h;
}

// argmax over V logits (first-max-wins, matching jnp.argmax). blockDim = 1024.
// step < 0: initial prediction (set wid/done). step >= 0: emit token with eos masking.
__global__ void argmax_kernel(const float* __restrict__ logits, int V, int step,
                              float* __restrict__ toks, int* __restrict__ wid,
                              int* __restrict__ done) {
    __shared__ float vals[1024];
    __shared__ int   idxs[1024];
    int tid = threadIdx.x;
    float best = -3.402823466e38f; int bi = 0x7fffffff;
    for (int v = tid; v < V; v += 1024) {
        float f = logits[v];
        if (f > best) { best = f; bi = v; }   // strict > keeps lowest index per thread
    }
    vals[tid] = best; idxs[tid] = bi;
    __syncthreads();
    for (int s = 512; s > 0; s >>= 1) {
        if (tid < s) {
            float ov = vals[tid + s]; int oi = idxs[tid + s];
            if (ov > vals[tid] || (ov == vals[tid] && oi < idxs[tid])) { vals[tid] = ov; idxs[tid] = oi; }
        }
        __syncthreads();
    }
    if (tid == 0) {
        int w = idxs[0];
        if (step < 0) { *wid = w; *done = (w == EOS_ID) ? 1 : 0; }
        else {
            int d = *done;
            int is_eos = (w == EOS_ID) ? 1 : 0;
            toks[step] = (float)((d || is_eos) ? -1 : w);
            *done = d | is_eos;
            *wid = w;
        }
    }
}

extern "C" void kernel_launch(void* const* d_in, const int* in_sizes, int n_in,
                              void* d_out, int out_size, void* d_ws, size_t ws_size,
                              hipStream_t stream) {
    const int*   src_ids      = (const int*)  d_in[0];
    const float* embed_input  = (const float*)d_in[1];
    const float* W_ih_e       = (const float*)d_in[2];
    const float* W_hh_e       = (const float*)d_in[3];
    const float* b_ih_e       = (const float*)d_in[4];
    const float* b_hh_e       = (const float*)d_in[5];
    const float* W_li         = (const float*)d_in[6];
    const float* b_li         = (const float*)d_in[7];
    const float* embed_target = (const float*)d_in[8];
    const float* W_ih_d       = (const float*)d_in[9];
    const float* W_hh_d       = (const float*)d_in[10];
    const float* b_ih_d       = (const float*)d_in[11];
    const float* b_hh_d       = (const float*)d_in[12];
    const float* W_lt         = (const float*)d_in[13];
    const float* b_lt         = (const float*)d_in[14];
    const float* W_tl         = (const float*)d_in[15];
    const float* b_tl         = (const float*)d_in[16];
    (void)in_sizes; (void)n_in; (void)out_size; (void)ws_size;

    // ---- workspace layout: bf16 weights + bf16 activations, then f32, then ints
    __bf16* bf = (__bf16*)d_ws;
    size_t o = 0;
    __bf16* Wihe_bf = bf + o; o += (size_t)H4DIM * HDIM;
    __bf16* Whhe_bf = bf + o; o += (size_t)H4DIM * HDIM;
    __bf16* Wihd_bf = bf + o; o += (size_t)H4DIM * HDIM;
    __bf16* Whhd_bf = bf + o; o += (size_t)H4DIM * HDIM;
    __bf16* Wtl_bf  = bf + o; o += (size_t)HDIM * 2 * HDIM;
    __bf16* Wli_bf  = bf + o; o += (size_t)VOUT * HDIM;
    __bf16* Wlt_bf  = bf + o; o += (size_t)VOUT * HDIM;
    __bf16* Xbf     = bf + o; o += (size_t)SLEN * HDIM;
    __bf16* hbf     = bf + o; o += HDIM;        // bf16 shadow of h / ht
    __bf16* catbf   = bf + o; o += 2 * HDIM;    // bf16 shadow of [ctx, ht]
    __bf16* htnbf   = bf + o; o += HDIM;        // bf16 shadow of ht_new
    __bf16* xrowbf  = bf + o; o += HDIM;        // bf16 gathered embed_target row
    o = (o + 1) & ~(size_t)1;                   // 4-byte align the f32 region
    float* fp = (float*)(bf + o);
    size_t fo = 0;
    float* G0      = fp + fo; fo += (size_t)SLEN * H4DIM;   // precomputed W_ih_e @ X
    float* hs_all  = fp + fo; fo += (size_t)SLEN * HDIM;
    float* hvec    = fp + fo; fo += HDIM;
    float* cvec    = fp + fo; fo += HDIM;
    float* gates   = fp + fo; fo += H4DIM;
    float* concat  = fp + fo; fo += 2 * HDIM;
    float* ht_new  = fp + fo; fo += HDIM;
    float* logits0 = fp + fo; fo += VOUT;
    int* wid  = (int*)(fp + fo);
    int* done = wid + 1;

    float* toks_out   = (float*)d_out;            // [51] emitted ids (as float)
    float* logits_out = (float*)d_out + NSTEPS;   // [51][32000]

    // ---- one-time (per launch) bf16 weight conversion
    cvt_f32_bf16_kernel<<<2048, 256, 0, stream>>>(W_ih_e, Wihe_bf, (size_t)H4DIM * HDIM);
    cvt_f32_bf16_kernel<<<2048, 256, 0, stream>>>(W_hh_e, Whhe_bf, (size_t)H4DIM * HDIM);
    cvt_f32_bf16_kernel<<<2048, 256, 0, stream>>>(W_ih_d, Wihd_bf, (size_t)H4DIM * HDIM);
    cvt_f32_bf16_kernel<<<2048, 256, 0, stream>>>(W_hh_d, Whhd_bf, (size_t)H4DIM * HDIM);
    cvt_f32_bf16_kernel<<<2048, 256, 0, stream>>>(W_tl,   Wtl_bf,  (size_t)HDIM * 2 * HDIM);
    cvt_f32_bf16_kernel<<<2048, 256, 0, stream>>>(W_li,   Wli_bf,  (size_t)VOUT * HDIM);
    cvt_f32_bf16_kernel<<<2048, 256, 0, stream>>>(W_lt,   Wlt_bf,  (size_t)VOUT * HDIM);

    init_state_kernel<<<4, 256, 0, stream>>>(hvec, cvec, hbf);
    embed_src_kernel<<<SLEN, 256, 0, stream>>>(src_ids, embed_input, Xbf);

    // ---- batched encoder input projection: G0[t][m] = (W_ih_e @ x_t)[m]
    wmma_gemm_nt_kernel<<<512, 128, 0, stream>>>(Wihe_bf, Xbf, G0, H4DIM, SLEN, HDIM);

    // ---- sequential encoder recurrence
    for (int t = 0; t < SLEN; ++t) {
        wmma_matvec_kernel<<<64, 128, 0, stream>>>(Whhe_bf, hbf, nullptr, nullptr,
            b_ih_e, b_hh_e, G0 + (size_t)t * H4DIM,
            gates, nullptr, H4DIM, HDIM, 0);
        lstm_cell_kernel<<<1, 1024, 0, stream>>>(gates, hvec, cvec, hbf,
                                                 hs_all + (size_t)t * HDIM);
    }

    // ---- first predicted token from final encoder state
    wmma_matvec_kernel<<<500, 128, 0, stream>>>(Wli_bf, hbf, nullptr, nullptr,
        b_li, nullptr, nullptr, logits0, nullptr, VOUT, HDIM, 0);
    argmax_kernel<<<1, 1024, 0, stream>>>(logits0, VOUT, -1, toks_out, wid, done);

    // ---- first decoder LSTM step: x = embed_target[wid0]
    gather_row_bf16_kernel<<<4, 256, 0, stream>>>(embed_target, wid, xrowbf);
    wmma_matvec_kernel<<<64, 128, 0, stream>>>(Wihd_bf, xrowbf, Whhd_bf, hbf,
        b_ih_d, b_hh_d, nullptr, gates, nullptr, H4DIM, HDIM, 0);
    lstm_cell_kernel<<<1, 1024, 0, stream>>>(gates, hvec, cvec, hbf, nullptr);

    // ---- greedy decode loop (51 fixed steps)
    for (int st = 0; st < NSTEPS; ++st) {
        attn_kernel<<<1, 1024, 0, stream>>>(hs_all, hvec, concat, catbf);
        // ht_new = tanh(W_tl @ [ctx, ht] + b_tl)   (K = 2048), bf16 shadow for W_lt
        wmma_matvec_kernel<<<16, 128, 0, stream>>>(Wtl_bf, catbf, nullptr, nullptr,
            b_tl, nullptr, nullptr, ht_new, htnbf, HDIM, 2 * HDIM, 1);
        // logits = W_lt @ ht_new + b_lt  (written straight into d_out)
        wmma_matvec_kernel<<<500, 128, 0, stream>>>(Wlt_bf, htnbf, nullptr, nullptr,
            b_lt, nullptr, nullptr, logits_out + (size_t)st * VOUT, nullptr, VOUT, HDIM, 0);
        // gather + LSTM gates must consume the CARRIED wid (before argmax updates it)
        gather_row_bf16_kernel<<<4, 256, 0, stream>>>(embed_target, wid, xrowbf);
        wmma_matvec_kernel<<<64, 128, 0, stream>>>(Wihd_bf, xrowbf, Whhd_bf, hbf,
            b_ih_d, b_hh_d, nullptr, gates, nullptr, H4DIM, HDIM, 0);
        // now update wid/done and emit the token
        argmax_kernel<<<1, 1024, 0, stream>>>(logits_out + (size_t)st * VOUT, VOUT, st,
                                              toks_out, wid, done);
        lstm_cell_kernel<<<1, 1024, 0, stream>>>(gates, hvec, cvec, hbf, nullptr);
    }
}
